// _SelfAttentionBlock_12567074308389
// MI455X (gfx1250) — matrix operand order repaired and checked by reading, not compiled, not run on hardware
//
#include <hip/hip_runtime.h>
#include <hip/hip_bf16.h>
#include <math.h>

typedef __attribute__((ext_vector_type(16))) _Float16 v16h;
typedef __attribute__((ext_vector_type(8)))  float    v8f;
typedef __attribute__((ext_vector_type(4)))  float    f32x4;

#define HWSZ 5184
#define HDIM 72
#define CIN  512
#define KC   256
#define VC   256
#define SLEN 110
#define SPAD 112   // padded row length for sim / keyt (16B-aligned f32x4 rows)
#define NB   8
#define LDA  40    // LDS row pitch in halfwords (80B: keeps b128 frags aligned)
#define LDB  40

union FragH { v16h v; uint4 q[2]; };

__device__ __forceinline__ f32x4 ld4(const float* __restrict__ p) {
  return *reinterpret_cast<const f32x4*>(p);
}
__device__ __forceinline__ f32x4 zero4() { return f32x4{0.f, 0.f, 0.f, 0.f}; }

// Single-instruction pack: v_cvt_pk_rtz_f16_f32
__device__ __forceinline__ unsigned pack2(float a, float b) {
  return __builtin_bit_cast(unsigned, __builtin_amdgcn_cvt_pkrtz(a, b));
}
__device__ __forceinline__ uint2 pack4(f32x4 v) {
  uint2 r;
  r.x = pack2(v[0], v[1]);
  r.y = pack2(v[2], v[3]);
  return r;
}

// One K=32 step: wave computes 16(M) x 64(N) via 4 WMMAs.
// All fragments are loaded into distinct registers first so the four WMMAs can
// issue back-to-back without per-WMMA s_wait_dscnt (WAR-free).
__device__ __forceinline__ void wave_tile_mma(
    const _Float16 (*As)[LDA], const _Float16 (*Bs)[LDB],
    int lane, int waveM, int waveN, v8f* acc)
{
  const int lr = lane & 15, lg = lane >> 4;
  FragH a, b[4];
  a.q[0] = *reinterpret_cast<const uint4*>(&As[waveM * 16 + lr][8 * lg]);
  a.q[1] = *reinterpret_cast<const uint4*>(&As[waveM * 16 + lr][16 + 8 * lg]);
  const _Float16* bp = &Bs[waveN * 64 + lr][8 * lg];
#pragma unroll
  for (int nt = 0; nt < 4; ++nt) {
    b[nt].q[0] = *reinterpret_cast<const uint4*>(bp);
    b[nt].q[1] = *reinterpret_cast<const uint4*>(bp + 16);
    bp += 16 * LDB;
  }
#pragma unroll
  for (int nt = 0; nt < 4; ++nt)
    acc[nt] = __builtin_amdgcn_wmma_f32_16x16x32_f16(
        false, a.v, false, b[nt].v, (short)0, acc[nt], false, false);
}

// ---------------- Kernel 1: fused k/v conv1x1 GEMM --------------------------
__global__ void __launch_bounds__(256) kv_gemm_kernel(
    const float* __restrict__ x,
    const float* __restrict__ Wk, const float* __restrict__ bk,
    const float* __restrict__ gamma, const float* __restrict__ beta,
    const float* __restrict__ rmean, const float* __restrict__ rvar,
    const float* __restrict__ Wv, const float* __restrict__ bv,
    float* __restrict__ kbuf, float* __restrict__ vbuf)
{
  __shared__ alignas(16) _Float16 As[2][64][LDA];
  __shared__ alignas(16) _Float16 Bs[2][128][LDB];
  const int t = threadIdx.x, lane = t & 31, wid = t >> 5;
  const int waveM = wid >> 1, waveN = wid & 1;
  const int b = blockIdx.z;
  const int blockM = blockIdx.y * 64;
  const int blockN = blockIdx.x * 128;
  const float* xb = x + (size_t)b * CIN * HWSZ;

  // A staging: 2 quads/thread along K (contiguous -> uint2 LDS stores)
  const int ar0 = t >> 3, ak = (t & 7) * 4;
  const int gm0 = blockM + ar0, gm1 = gm0 + 32;
  const float* ap0 = ((gm0 < KC) ? (Wk + (size_t)gm0 * CIN)
                                 : (Wv + (size_t)(gm0 - KC) * CIN)) + ak;
  const float* ap1 = ((gm1 < KC) ? (Wk + (size_t)gm1 * CIN)
                                 : (Wv + (size_t)(gm1 - KC) * CIN)) + ak;
  // B staging: pair-of-K-rows trick -> packed b32 transposed stores.
  // Edge columns: clamp pointer in-bounds and load anyway; garbage columns only
  // feed accumulator outputs that the epilogue never stores.
  const int bn4 = (t & 31) * 4, bkr = (t >> 5) * 2;
  const int gbn = (blockN + bn4 < HWSZ) ? (blockN + bn4) : (HWSZ - 4);
  const float* bp = xb + (size_t)bkr * HWSZ + gbn;

  uint2 pa0, pa1; unsigned pb[8];
  auto load_tiles = [&]() {
    pa0 = pack4(ld4(ap0)); pa1 = pack4(ld4(ap1));
    f32x4 r00 = ld4(bp),             r01 = ld4(bp + HWSZ);
    f32x4 r10 = ld4(bp + 16 * HWSZ), r11 = ld4(bp + 17 * HWSZ);
#pragma unroll
    for (int j = 0; j < 4; ++j) {
      pb[j]     = pack2(r00[j], r01[j]);
      pb[4 + j] = pack2(r10[j], r11[j]);
    }
    ap0 += 32; ap1 += 32; bp += (size_t)32 * HWSZ;
  };
  auto store_tiles = [&](int buf) {
    *reinterpret_cast<uint2*>(&As[buf][ar0][ak])      = pa0;
    *reinterpret_cast<uint2*>(&As[buf][ar0 + 32][ak]) = pa1;
#pragma unroll
    for (int j = 0; j < 4; ++j) {
      *reinterpret_cast<unsigned*>(&Bs[buf][bn4 + j][bkr])      = pb[j];
      *reinterpret_cast<unsigned*>(&Bs[buf][bn4 + j][bkr + 16]) = pb[4 + j];
    }
  };

  v8f acc[4] = {};
  load_tiles(); store_tiles(0); __syncthreads();
  const int NS = CIN / 32;
#pragma unroll 2
  for (int s = 0; s < NS; ++s) {
    if (s + 1 < NS) load_tiles();
    wave_tile_mma(As[s & 1], Bs[s & 1], lane, waveM, waveN, acc);
    if (s + 1 < NS) { store_tiles((s + 1) & 1); __syncthreads(); }
  }

  const int lr = lane & 15, lg = lane >> 4;
#pragma unroll
  for (int r = 0; r < 8; ++r) {
    int gm = blockM + waveM * 16 + r + 8 * lg;
    if (gm < KC) {
      float inv = gamma[gm] * rsqrtf(rvar[gm] + 1e-5f);
      float off = (bk[gm] - rmean[gm]) * inv + beta[gm];
      float* dst = kbuf + ((size_t)b * KC + gm) * HWSZ;
#pragma unroll
      for (int nt = 0; nt < 4; ++nt) {
        int gn = blockN + waveN * 64 + nt * 16 + lr;
        if (gn < HWSZ) dst[gn] = fmaxf(acc[nt][r] * inv + off, 0.f);
      }
    } else {
      float bias = bv[gm - KC];
      float* dst = vbuf + ((size_t)b * VC + gm - KC) * HWSZ;
#pragma unroll
      for (int nt = 0; nt < 4; ++nt) {
        int gn = blockN + waveN * 64 + nt * 16 + lr;
        if (gn < HWSZ) dst[gn] = acc[nt][r] + bias;
      }
    }
  }
}

// ---------------- Kernel 2: PSP pooling (1,3,6,8) ---------------------------
__global__ void __launch_bounds__(256) psp_pool_kernel(
    const float* __restrict__ in, float* __restrict__ out, int transpose)
{
  __shared__ float pl[HWSZ];
  __shared__ float s8[64], s6[36], s3[9], s1v;
  const int t = threadIdx.x;
  const float* plane = in + (size_t)blockIdx.x * HWSZ;
  for (int i = 0; i < 21; ++i) {
    int idx = t + 256 * i;
    if (idx < HWSZ) pl[idx] = plane[idx];
  }
  __syncthreads();
  if (t < 64) {
    int i = t >> 3, j = t & 7;
    float s = 0.f;
    for (int r = 0; r < 9; ++r)
      for (int c = 0; c < 9; ++c) s += pl[(i * 9 + r) * HDIM + j * 9 + c];
    s8[t] = s * (1.f / 81.f);
  } else if (t < 100) {
    int u = t - 64, i = u / 6, j = u % 6;
    float s = 0.f;
    for (int r = 0; r < 12; ++r)
      for (int c = 0; c < 12; ++c) s += pl[(i * 12 + r) * HDIM + j * 12 + c];
    s6[u] = s * (1.f / 144.f);
  }
  __syncthreads();
  if (t < 9) {
    int i = t / 3, j = t % 3;
    s3[t] = 0.25f * (s6[(2 * i) * 6 + 2 * j] + s6[(2 * i) * 6 + 2 * j + 1] +
                     s6[(2 * i + 1) * 6 + 2 * j] + s6[(2 * i + 1) * 6 + 2 * j + 1]);
  } else if (t == 9) {
    float s = 0.f;
    for (int i = 0; i < 64; ++i) s += s8[i];
    s1v = s * (1.f / 64.f);
  }
  __syncthreads();
  int b = blockIdx.x >> 8;
  int c = blockIdx.x & 255;
  if (t < SPAD) {
    float val = (t >= SLEN) ? 0.f
              : (t == 0)  ? s1v
              : (t < 10)  ? s3[t - 1]
              : (t < 46)  ? s6[t - 10] : s8[t - 46];
    if (!transpose) out[((size_t)b * KC + c) * SPAD + t] = val;     // keyt padded
    else if (t < SLEN) out[((size_t)b * SLEN + t) * VC + c] = val;  // value
  }
}

// ---------------- Kernel 3: sim = query @ keyt * scale ----------------------
__global__ void __launch_bounds__(256) sim_gemm_kernel(
    const float* __restrict__ kbuf, const float* __restrict__ keyt,
    float* __restrict__ sim)
{
  __shared__ alignas(16) _Float16 As[2][64][LDA];
  __shared__ alignas(16) _Float16 Bs[2][128][LDB];
  const int t = threadIdx.x, lane = t & 31, wid = t >> 5;
  const int waveM = wid >> 1, waveN = wid & 1;
  const int b = blockIdx.z;
  const int blockM = blockIdx.y * 64;

  // A: query (transpose of kbuf layout), pair-of-K-rows
  const int am4 = (t & 15) * 4, akr = (t >> 4) * 2;
  const float* ap = kbuf + ((size_t)b * KC + akr) * HWSZ + blockM + am4;
  // B: keyt (padded rows), pair-of-K-rows. Columns >=110 produce garbage that
  // only lands in discarded outputs (reads stay inside the workspace).
  const int bn4 = (t & 31) * 4, bkr = (t >> 5) * 2;
  const float* bp = keyt + ((size_t)b * KC + bkr) * SPAD + bn4;

  unsigned pa[4], pb[8];
  auto load_tiles = [&]() {
    f32x4 a0 = ld4(ap), a1 = ld4(ap + HWSZ);
#pragma unroll
    for (int j = 0; j < 4; ++j) pa[j] = pack2(a0[j], a1[j]);
    f32x4 r00 = ld4(bp),             r01 = ld4(bp + SPAD);
    f32x4 r10 = ld4(bp + 16 * SPAD), r11 = ld4(bp + 17 * SPAD);
#pragma unroll
    for (int j = 0; j < 4; ++j) {
      pb[j]     = pack2(r00[j], r01[j]);
      pb[4 + j] = pack2(r10[j], r11[j]);
    }
    ap += (size_t)32 * HWSZ; bp += 32 * SPAD;
  };
  auto store_tiles = [&](int buf) {
#pragma unroll
    for (int j = 0; j < 4; ++j)
      *reinterpret_cast<unsigned*>(&As[buf][am4 + j][akr]) = pa[j];
#pragma unroll
    for (int j = 0; j < 4; ++j) {
      *reinterpret_cast<unsigned*>(&Bs[buf][bn4 + j][bkr])      = pb[j];
      *reinterpret_cast<unsigned*>(&Bs[buf][bn4 + j][bkr + 16]) = pb[4 + j];
    }
  };

  v8f acc[4] = {};
  load_tiles(); store_tiles(0); __syncthreads();
  const int NS = KC / 32;
#pragma unroll 2
  for (int s = 0; s < NS; ++s) {
    if (s + 1 < NS) load_tiles();
    wave_tile_mma(As[s & 1], Bs[s & 1], lane, waveM, waveN, acc);
    if (s + 1 < NS) { store_tiles((s + 1) & 1); __syncthreads(); }
  }

  const int lr = lane & 15, lg = lane >> 4;
#pragma unroll
  for (int r = 0; r < 8; ++r) {
    int gm = blockM + waveM * 16 + r + 8 * lg;
#pragma unroll
    for (int nt = 0; nt < 4; ++nt) {
      int gn = waveN * 64 + nt * 16 + lr;
      if (gn < SLEN)
        sim[((size_t)b * HWSZ + gm) * SPAD + gn] = acc[nt][r] * 0.0625f;
    }
  }
}

// ---------------- Kernel 4: softmax over S=110 ------------------------------
__global__ void __launch_bounds__(256) softmax_kernel(float* __restrict__ sim)
{
  const int lane = threadIdx.x & 31, wid = threadIdx.x >> 5;
  const size_t row = (size_t)blockIdx.x * 8 + wid;
  float* p = sim + row * SPAD;
  float v[4];
  float m = -1e30f;
#pragma unroll
  for (int i = 0; i < 4; ++i) {
    int j = lane + 32 * i;
    v[i] = (j < SLEN) ? p[j] : -1e30f;
    m = fmaxf(m, v[i]);
  }
#pragma unroll
  for (int off = 16; off > 0; off >>= 1) m = fmaxf(m, __shfl_xor(m, off, 32));
  float s = 0.f;
#pragma unroll
  for (int i = 0; i < 4; ++i) { v[i] = __expf(v[i] - m); s += v[i]; }
#pragma unroll
  for (int off = 16; off > 0; off >>= 1) s += __shfl_xor(s, off, 32);
  float inv = 1.f / s;
#pragma unroll
  for (int i = 0; i < 4; ++i) {
    int j = lane + 32 * i;
    if (j < SLEN) p[j] = v[i] * inv;
  }
}

// ---------------- Kernel 5: context = P @ value (K padded 110->128) ---------
__global__ void __launch_bounds__(256) ctx_gemm_kernel(
    const float* __restrict__ sim, const float* __restrict__ value,
    float* __restrict__ ctx)
{
  __shared__ alignas(16) _Float16 As[2][64][LDA];
  __shared__ alignas(16) _Float16 Bs[2][128][LDB];
  const int t = threadIdx.x, lane = t & 31, wid = t >> 5;
  const int waveM = wid >> 1, waveN = wid & 1;
  const int b = blockIdx.z;
  const int blockM = blockIdx.y * 64;
  const int blockN = blockIdx.x * 128;

  // A: P rows (K contiguous) -> uint2 stores; K >= 110 must be zero (K-padding
  // pollutes every output, so these guards stay; loop is fully unrolled so they
  // only materialize in the last step).
  const int ar0 = t >> 3, ak = (t & 7) * 4;
  const float* ap0 = sim + ((size_t)b * HWSZ + blockM + ar0) * SPAD + ak;
  const float* ap1 = ap0 + (size_t)32 * SPAD;
  // B: value rows, pair-of-K-rows, K-guarded
  const int bn4 = (t & 31) * 4, bkr = (t >> 5) * 2;
  const float* bp = value + ((size_t)b * SLEN + bkr) * VC + blockN + bn4;

  int know = 0;
  uint2 pa0, pa1; unsigned pb[8];
  auto load_tiles = [&]() {
    f32x4 a0 = ld4(ap0), a1 = ld4(ap1);
#pragma unroll
    for (int j = 0; j < 4; ++j)
      if (know + ak + j >= SLEN) { a0[j] = 0.f; a1[j] = 0.f; }
    pa0 = pack4(a0); pa1 = pack4(a1);
    f32x4 r00 = zero4(), r01 = zero4(), r10 = zero4(), r11 = zero4();
    if (know + bkr      < SLEN) r00 = ld4(bp);
    if (know + bkr + 1  < SLEN) r01 = ld4(bp + VC);
    if (know + bkr + 16 < SLEN) r10 = ld4(bp + 16 * VC);
    if (know + bkr + 17 < SLEN) r11 = ld4(bp + 17 * VC);
#pragma unroll
    for (int j = 0; j < 4; ++j) {
      pb[j]     = pack2(r00[j], r01[j]);
      pb[4 + j] = pack2(r10[j], r11[j]);
    }
    ap0 += 32; ap1 += 32; bp += (size_t)32 * VC; know += 32;
  };
  auto store_tiles = [&](int buf) {
    *reinterpret_cast<uint2*>(&As[buf][ar0][ak])      = pa0;
    *reinterpret_cast<uint2*>(&As[buf][ar0 + 32][ak]) = pa1;
#pragma unroll
    for (int j = 0; j < 4; ++j) {
      *reinterpret_cast<unsigned*>(&Bs[buf][bn4 + j][bkr])      = pb[j];
      *reinterpret_cast<unsigned*>(&Bs[buf][bn4 + j][bkr + 16]) = pb[4 + j];
    }
  };

  v8f acc[4] = {};
  load_tiles(); store_tiles(0); __syncthreads();
  const int NS = 4;  // K = 128
#pragma unroll
  for (int s = 0; s < NS; ++s) {
    if (s + 1 < NS) load_tiles();
    wave_tile_mma(As[s & 1], Bs[s & 1], lane, waveM, waveN, acc);
    if (s + 1 < NS) { store_tiles((s + 1) & 1); __syncthreads(); }
  }

  const int lr = lane & 15, lg = lane >> 4;
#pragma unroll
  for (int r = 0; r < 8; ++r) {
    int gm = blockM + waveM * 16 + r + 8 * lg;
#pragma unroll
    for (int nt = 0; nt < 4; ++nt) {
      int gn = blockN + waveN * 64 + nt * 16 + lr;
      ctx[((size_t)b * HWSZ + gm) * VC + gn] = acc[nt][r];
    }
  }
}

// ---------------- Kernel 6: out = Ww @ context^T + bw -----------------------
__global__ void __launch_bounds__(256) out_gemm_kernel(
    const float* __restrict__ ctx, const float* __restrict__ Ww,
    const float* __restrict__ bw, float* __restrict__ out)
{
  __shared__ alignas(16) _Float16 As[2][64][LDA];
  __shared__ alignas(16) _Float16 Bs[2][128][LDB];
  const int t = threadIdx.x, lane = t & 31, wid = t >> 5;
  const int waveM = wid >> 1, waveN = wid & 1;
  const int b = blockIdx.z;
  const int blockM = blockIdx.y * 64;
  const int blockN = blockIdx.x * 128;

  const int ar0 = t >> 3, ak = (t & 7) * 4;
  const float* ap0 = Ww + (size_t)(blockM + ar0) * VC + ak;
  const float* ap1 = ap0 + (size_t)32 * VC;
  // B: ctx is [pixel][vc]; clamp edge pixels in-bounds and always load —
  // garbage columns only reach discarded outputs.
  const int bn0 = t >> 3, bk = (t & 7) * 4;
  const float* bp[4];
#pragma unroll
  for (int i = 0; i < 4; ++i) {
    int gn = blockN + bn0 + 32 * i;
    int gnc = (gn < HWSZ) ? gn : (HWSZ - 1);
    bp[i] = ctx + ((size_t)b * HWSZ + gnc) * VC + bk;
  }

  uint2 pa0, pa1, pbq[4];
  auto load_tiles = [&]() {
    pa0 = pack4(ld4(ap0)); pa1 = pack4(ld4(ap1));
#pragma unroll
    for (int i = 0; i < 4; ++i) {
      pbq[i] = pack4(ld4(bp[i]));
      bp[i] += 32;
    }
    ap0 += 32; ap1 += 32;
  };
  auto store_tiles = [&](int buf) {
    *reinterpret_cast<uint2*>(&As[buf][ar0][ak])      = pa0;
    *reinterpret_cast<uint2*>(&As[buf][ar0 + 32][ak]) = pa1;
#pragma unroll
    for (int i = 0; i < 4; ++i)
      *reinterpret_cast<uint2*>(&Bs[buf][bn0 + 32 * i][bk]) = pbq[i];
  };

  v8f acc[4] = {};
  load_tiles(); store_tiles(0); __syncthreads();
  const int NS = VC / 32;
#pragma unroll 2
  for (int s = 0; s < NS; ++s) {
    if (s + 1 < NS) load_tiles();
    wave_tile_mma(As[s & 1], Bs[s & 1], lane, waveM, waveN, acc);
    if (s + 1 < NS) { store_tiles((s + 1) & 1); __syncthreads(); }
  }

  const int lr = lane & 15, lg = lane >> 4;
#pragma unroll
  for (int r = 0; r < 8; ++r) {
    int gm = blockM + waveM * 16 + r + 8 * lg;
    float bias = bw[gm];
#pragma unroll
    for (int nt = 0; nt < 4; ++nt) {
      int gn = blockN + waveN * 64 + nt * 16 + lr;
      if (gn < HWSZ)
        out[((size_t)b * CIN + gm) * HWSZ + gn] = acc[nt][r] + bias;
    }
  }
}

extern "C" void kernel_launch(void* const* d_in, const int* in_sizes, int n_in,
                              void* d_out, int out_size, void* d_ws, size_t ws_size,
                              hipStream_t stream) {
  (void)in_sizes; (void)n_in; (void)out_size; (void)ws_size;
  const float* x     = (const float*)d_in[0];
  const float* Wk    = (const float*)d_in[1];
  const float* bk    = (const float*)d_in[2];
  const float* gamma = (const float*)d_in[3];
  const float* beta  = (const float*)d_in[4];
  const float* rmean = (const float*)d_in[5];
  const float* rvar  = (const float*)d_in[6];
  const float* Wv    = (const float*)d_in[7];
  const float* bv    = (const float*)d_in[8];
  const float* Ww    = (const float*)d_in[9];
  const float* bw    = (const float*)d_in[10];
  float* out = (float*)d_out;

  // Workspace (floats), ~87 MB with aliasing:
  //   kbuf [B,256,HW]   -> reused as context [B,HW,256]
  //   vbuf [B,256,HW]   -> reused as sim     [B,HW,112]
  //   keyt [B,256,112], value [B,110,256]
  float* ws = (float*)d_ws;
  const size_t KBUF = (size_t)NB * KC * HWSZ;
  float* kbuf  = ws;
  float* vbuf  = ws + KBUF;
  float* keyt  = ws + 2 * KBUF;
  float* value = keyt + (size_t)NB * KC * SPAD;
  float* sim   = vbuf;
  float* ctx   = kbuf;

  kv_gemm_kernel<<<dim3(41, 8, NB), 256, 0, stream>>>(
      x, Wk, bk, gamma, beta, rmean, rvar, Wv, bv, kbuf, vbuf);
  psp_pool_kernel<<<NB * KC, 256, 0, stream>>>(kbuf, keyt, 0);
  psp_pool_kernel<<<NB * VC, 256, 0, stream>>>(vbuf, value, 1);
  sim_gemm_kernel<<<dim3(1, 81, NB), 256, 0, stream>>>(kbuf, keyt, sim);
  softmax_kernel<<<HWSZ, 256, 0, stream>>>(sim);
  ctx_gemm_kernel<<<dim3(2, 81, NB), 256, 0, stream>>>(sim, value, ctx);
  out_gemm_kernel<<<dim3(41, 8, NB), 256, 0, stream>>>(ctx, Ww, bw, out);
}
